// SelfAttention_48344151884270
// MI455X (gfx1250) — matrix operand order repaired and checked by reading, not compiled, not run on hardware
//
#include <hip/hip_runtime.h>

#define B_ 2
#define N_ 2048
#define C_ 1024
#define H_ 16
#define D_ 64
#define M1 (B_ * N_)   // 4096 rows
#define N3 (3 * C_)    // 3072 qkv cols

typedef __attribute__((ext_vector_type(16))) __bf16 v16bf;
typedef __attribute__((ext_vector_type(8)))  __bf16 v8bf;
typedef __attribute__((ext_vector_type(8)))  float  v8f;
typedef __attribute__((ext_vector_type(4)))  unsigned int v4u;
typedef __attribute__((ext_vector_type(8)))  int v8i;
typedef __attribute__((ext_vector_type(4)))  int v4i;

union Frag16 { v16bf v; v8bf h[2]; };

__device__ __forceinline__ unsigned short f2bfu(float f) {
  unsigned u = __float_as_uint(f);
  u += 0x7fffu + ((u >> 16) & 1u);   // round-to-nearest-even
  return (unsigned short)(u >> 16);
}
__device__ __forceinline__ __bf16 f2bf(float f) {
  unsigned short s = f2bfu(f);
  return __builtin_bit_cast(__bf16, s);
}

__device__ __forceinline__ v8f vzero() {
  v8f z;
#pragma unroll
  for (int i = 0; i < 8; ++i) z[i] = 0.f;
  return z;
}

// A-fragment (16x32, 16-bit): lane covers row (lane&15); contiguous K=[16*hi,+16)
__device__ __forceinline__ v16bf load_a_frag(const __bf16* p, int hi) {
  Frag16 f;
  f.h[0] = *(const v8bf*)(p + 16 * hi);
  f.h[1] = *(const v8bf*)(p + 16 * hi + 8);
  return f.v;
}
// B-fragment (32x16, 16-bit): lane holds col (lane&15), contiguous K=[16*hi,+16) at p
__device__ __forceinline__ v16bf load_b_frag(const __bf16* p) {
  Frag16 f;
  f.h[0] = *(const v8bf*)(p);
  f.h[1] = *(const v8bf*)(p + 8);
  return f.v;
}

__device__ __forceinline__ v8f wmma_bf16(v16bf a, v16bf b, v8f c) {
  return __builtin_amdgcn_wmma_f32_16x16x32_bf16(false, a, false, b, (short)0, c,
                                                 false, false);
}

// ---- Tensor Data Mover: 2-D tile load (bf16 elements) per CDNA5 ISA ----
// tile_d0 = contiguous extent (elements), tile_d1 = rows; stride0 in elements.
// 6-arg builtin form (clang-23 toolchain): (g0, g1, g2, g3, extra, cpol).
__device__ __forceinline__ void tdm_load_2d(unsigned lds_addr, const void* gaddr,
                                            unsigned tile_d0, unsigned tile_d1,
                                            unsigned tdim0, unsigned tdim1,
                                            unsigned long long stride0) {
  unsigned long long ga = (unsigned long long)(uintptr_t)gaddr;
  v4u g0;
  g0[0] = 1u;                                         // count=1 (user descriptor)
  g0[1] = lds_addr;                                   // lds_addr [63:32]
  g0[2] = (unsigned)(ga & 0xffffffffu);               // global_addr lo
  g0[3] = (unsigned)((ga >> 32) & 0x01ffffffu) | (2u << 30);  // addr hi | type=2
  v8i g1;
  g1[0] = (int)(1u << 16);                            // data_size=1 (2 bytes), wg_mask=0
  g1[1] = (int)((tdim0 & 0xffffu) << 16);             // tensor_dim0[15:0] @bits63:48
  g1[2] = (int)(((tdim0 >> 16) & 0xffffu) | ((tdim1 & 0xffffu) << 16));
  g1[3] = (int)(((tdim1 >> 16) & 0xffffu) | ((tile_d0 & 0xffffu) << 16));
  g1[4] = (int)(tile_d1 & 0xffffu);                   // tile_dim1; tile_dim2=0
  g1[5] = (int)(stride0 & 0xffffffffu);               // tensor_dim0_stride lo
  g1[6] = (int)((stride0 >> 32) & 0xffffu);           // stride0 hi; stride1=0
  g1[7] = 0;
  v4i gz4 = {0, 0, 0, 0};
  v8i gz8 = {0, 0, 0, 0, 0, 0, 0, 0};
  __builtin_amdgcn_tensor_load_to_lds(g0, g1, gz4, gz4, gz8, 0);
}

// ---------------- conversion kernels ----------------
__global__ void cvt_bf16(const float* __restrict__ s, unsigned short* __restrict__ d, int n) {
  int i = blockIdx.x * blockDim.x + threadIdx.x;
  if (i < n) d[i] = f2bfu(s[i]);
}

// s is [R x Cc] row-major; d becomes [Cc x R] row-major (transposed)
__global__ void cvt_bf16_T(const float* __restrict__ s, unsigned short* __restrict__ d,
                           int R, int Cc) {
  int i = blockIdx.x * blockDim.x + threadIdx.x;
  if (i < R * Cc) {
    int r = i % R;
    int c = i / R;
    d[(size_t)c * R + r] = f2bfu(s[(size_t)r * Cc + c]);
  }
}

// ---------------- WMMA GEMM: C[M,N] = A[M,K] * BT[N,K]^T (+bias), pipelined ----------------
__global__ __launch_bounds__(256) void gemm_bf16(const unsigned short* __restrict__ Au,
                                                 const unsigned short* __restrict__ BTu,
                                                 float* __restrict__ Cm,
                                                 const float* __restrict__ bias,
                                                 int M, int Nn, int K) {
  const __bf16* A  = reinterpret_cast<const __bf16*>(Au);
  const __bf16* BT = reinterpret_cast<const __bf16*>(BTu);
  int lane = threadIdx.x & 31;
  int wave = threadIdx.x >> 5;
  int hi   = lane >> 4;
  int ln   = lane & 15;
  int m0 = blockIdx.y * 128 + wave * 16;
  int n0 = blockIdx.x * 64;
  if (m0 >= M) return;

  const __bf16* arow = A + (size_t)(m0 + ln) * K;
  const __bf16* bcol = BT + (size_t)(n0 + ln) * K + 16 * hi;
  size_t bstep = (size_t)16 * K;

  v8f acc[4];
#pragma unroll
  for (int t = 0; t < 4; ++t) acc[t] = vzero();

  // software pipeline: fragments for step k held while step k+32 loads are in flight
  v16bf a_c  = load_a_frag(arow, hi);
  v16bf b_c0 = load_b_frag(bcol);
  v16bf b_c1 = load_b_frag(bcol + bstep);
  v16bf b_c2 = load_b_frag(bcol + 2 * bstep);
  v16bf b_c3 = load_b_frag(bcol + 3 * bstep);

  for (int k = 32; k < K; k += 32) {
    v16bf a_n  = load_a_frag(arow + k, hi);
    v16bf b_n0 = load_b_frag(bcol + k);
    v16bf b_n1 = load_b_frag(bcol + bstep + k);
    v16bf b_n2 = load_b_frag(bcol + 2 * bstep + k);
    v16bf b_n3 = load_b_frag(bcol + 3 * bstep + k);
    acc[0] = wmma_bf16(a_c, b_c0, acc[0]);
    acc[1] = wmma_bf16(a_c, b_c1, acc[1]);
    acc[2] = wmma_bf16(a_c, b_c2, acc[2]);
    acc[3] = wmma_bf16(a_c, b_c3, acc[3]);
    a_c = a_n; b_c0 = b_n0; b_c1 = b_n1; b_c2 = b_n2; b_c3 = b_n3;
  }
  acc[0] = wmma_bf16(a_c, b_c0, acc[0]);
  acc[1] = wmma_bf16(a_c, b_c1, acc[1]);
  acc[2] = wmma_bf16(a_c, b_c2, acc[2]);
  acc[3] = wmma_bf16(a_c, b_c3, acc[3]);

#pragma unroll
  for (int t = 0; t < 4; ++t) {
    int col = n0 + 16 * t + ln;
    float bv = bias ? bias[col] : 0.f;
#pragma unroll
    for (int r = 0; r < 8; ++r) {
      int row = m0 + r + 8 * hi;
      Cm[(size_t)row * Nn + col] = acc[t][r] + bv;
    }
  }
}

// ---------------- QKV prep: per-head LN(q)*8, LN(k), V -> [B,H,D,N] bf16 ----------------
__global__ void qkv_prep(const float* __restrict__ qkv,
                         const float* __restrict__ lnw, const float* __restrict__ lnb,
                         unsigned short* __restrict__ qb, unsigned short* __restrict__ kb,
                         unsigned short* __restrict__ vT) {
  int idx = blockIdx.x * blockDim.x + threadIdx.x;
  if (idx >= B_ * N_ * H_) return;
  int h = idx % H_;
  int n = (idx / H_) % N_;
  int b = idx / (H_ * N_);
  const float* row = qkv + (size_t)(b * N_ + n) * N3;
  const float* qv = row + h * D_;
  const float* kv = row + C_ + h * D_;
  const float* vv = row + 2 * C_ + h * D_;
  size_t hb = ((size_t)(b * H_ + h) * N_ + n) * D_;

  float s = 0.f, ss = 0.f;
  for (int d = 0; d < D_; ++d) { float x = qv[d]; s += x; ss += x * x; }
  float mu = s / D_;
  float rs = rsqrtf(ss / D_ - mu * mu + 1e-5f);
  for (int d = 0; d < D_; ++d)
    qb[hb + d] = f2bfu(((qv[d] - mu) * rs * lnw[d] + lnb[d]) * 8.0f);  // * sqrt(64)

  s = 0.f; ss = 0.f;
  for (int d = 0; d < D_; ++d) { float x = kv[d]; s += x; ss += x * x; }
  mu = s / D_;
  rs = rsqrtf(ss / D_ - mu * mu + 1e-5f);
  for (int d = 0; d < D_; ++d)
    kb[hb + d] = f2bfu((kv[d] - mu) * rs * lnw[d] + lnb[d]);

  size_t vb = (size_t)(b * H_ + h) * D_ * N_;
  for (int d = 0; d < D_; ++d)
    vT[vb + (size_t)d * N_ + n] = f2bfu(vv[d]);
}

// ---------------- causal flash attention: TDM-staged K/V, 4 waves share tiles ----------------
__global__ __launch_bounds__(128) void attn_fwd(const unsigned short* __restrict__ qu,
                                                const unsigned short* __restrict__ ku,
                                                const unsigned short* __restrict__ vu,
                                                unsigned short* __restrict__ ou) {
  const __bf16* Q  = reinterpret_cast<const __bf16*>(qu);
  const __bf16* Kk = reinterpret_cast<const __bf16*>(ku);
  const __bf16* Vt = reinterpret_cast<const __bf16*>(vu);
  __bf16* O = reinterpret_cast<__bf16*>(ou);

  __shared__ unsigned short kt_s[2][32 * 64];  // [key][d]  double-buffered, 2x4KB
  __shared__ unsigned short vt_s[2][64 * 32];  // [d][key]  double-buffered, 2x4KB
  __shared__ __bf16 pbuf[4][16 * 32];          // per-wave P transpose tiles

  int lane = threadIdx.x & 31;
  int wave = threadIdx.x >> 5;
  int hi = lane >> 4;
  int ln = lane & 15;
  int bh = blockIdx.x;
  int b = bh / H_, h = bh % H_;
  int qt = blockIdx.y * 64 + wave * 16;

  const __bf16* qp = Q + ((size_t)bh * N_ + qt) * D_;
  const __bf16* kp = Kk + (size_t)bh * N_ * D_;
  const __bf16* vp = Vt + (size_t)bh * D_ * N_;
  __bf16* pl = &pbuf[wave][0];

  v16bf aq0 = load_a_frag(qp + (size_t)ln * D_, hi);       // d 0..31
  v16bf aq1 = load_a_frag(qp + (size_t)ln * D_ + 32, hi);  // d 32..63

  v8f o0 = vzero(), o1 = vzero(), o2 = vzero(), o3 = vzero();
  float mr[8], lr[8];
#pragma unroll
  for (int r = 0; r < 8; ++r) { mr[r] = -3.0e38f; lr[r] = 0.f; }

  // uniform trip count across the block (extra blocks fully masked -> p == 0)
  int nkb = blockIdx.y * 2 + 2;

  if (wave == 0) {  // stage block 0 (TDM, tracked by TENSORcnt)
    tdm_load_2d((unsigned)(uintptr_t)&kt_s[0][0], kp, 64, 32, 64, N_, 64);
    tdm_load_2d((unsigned)(uintptr_t)&vt_s[0][0], vp, 32, 64, N_, 64, N_);
    __builtin_amdgcn_s_wait_tensorcnt(0);
  }
  __syncthreads();

  for (int kbI = 0; kbI < nkb; ++kbI) {
    int k0 = kbI * 32;
    int buf = kbI & 1;
    if (wave == 0 && kbI + 1 < nkb) {  // prefetch next tile into other buffer
      int k1 = k0 + 32;
      tdm_load_2d((unsigned)(uintptr_t)&kt_s[buf ^ 1][0], kp + (size_t)k1 * D_,
                  64, 32, 64, N_, 64);
      tdm_load_2d((unsigned)(uintptr_t)&vt_s[buf ^ 1][0], vp + k1,
                  32, 64, N_, 64, N_);
    }
    const __bf16* kt = (const __bf16*)&kt_s[buf][0];
    const __bf16* vt = (const __bf16*)&vt_s[buf][0];

    v8f s0 = vzero(), s1 = vzero();
    const __bf16* kc0 = kt + ln * 64 + 16 * hi;
    s0 = wmma_bf16(aq0, load_b_frag(kc0), s0);
    s0 = wmma_bf16(aq1, load_b_frag(kc0 + 32), s0);
    const __bf16* kc1 = kt + (16 + ln) * 64 + 16 * hi;
    s1 = wmma_bf16(aq0, load_b_frag(kc1), s1);
    s1 = wmma_bf16(aq1, load_b_frag(kc1 + 32), s1);

    if (k0 + 31 > qt) {  // diagonal / fully-masked blocks
#pragma unroll
      for (int r = 0; r < 8; ++r) {
        int row = qt + r + 8 * hi;
        if (k0 + ln > row)      s0[r] = -1.0e30f;
        if (k0 + 16 + ln > row) s1[r] = -1.0e30f;
      }
    }
    // online softmax; row (r + 8*hi) lives across this lane's 16-lane half
#pragma unroll
    for (int r = 0; r < 8; ++r) {
      float rm = fmaxf(s0[r], s1[r]);
      rm = fmaxf(rm, __shfl_xor(rm, 1));
      rm = fmaxf(rm, __shfl_xor(rm, 2));
      rm = fmaxf(rm, __shfl_xor(rm, 4));
      rm = fmaxf(rm, __shfl_xor(rm, 8));
      float mn = fmaxf(mr[r], rm);
      float alpha = __expf(mr[r] - mn);
      mr[r] = mn;
      float p0 = __expf(s0[r] - mn);
      float p1 = __expf(s1[r] - mn);
      s0[r] = p0; s1[r] = p1;
      float rsum = p0 + p1;
      rsum += __shfl_xor(rsum, 1);
      rsum += __shfl_xor(rsum, 2);
      rsum += __shfl_xor(rsum, 4);
      rsum += __shfl_xor(rsum, 8);
      lr[r] = lr[r] * alpha + rsum;
      o0[r] *= alpha; o1[r] *= alpha; o2[r] *= alpha; o3[r] *= alpha;
    }
    // C-layout -> A-layout via per-wave LDS tile
#pragma unroll
    for (int r = 0; r < 8; ++r) {
      int row = r + 8 * hi;
      pl[row * 32 + ln]      = f2bf(s0[r]);
      pl[row * 32 + 16 + ln] = f2bf(s1[r]);
    }
    asm volatile("s_wait_dscnt 0x0" ::: "memory");  // per-wave DS ordering
    v16bf ap = load_a_frag(pl + ln * 32, hi);

    const __bf16* vb0 = vt + ln * 32 + 16 * hi;
    o0 = wmma_bf16(ap, load_b_frag(vb0), o0);
    o1 = wmma_bf16(ap, load_b_frag(vb0 + 16 * 32), o1);
    o2 = wmma_bf16(ap, load_b_frag(vb0 + 32 * 32), o2);
    o3 = wmma_bf16(ap, load_b_frag(vb0 + 48 * 32), o3);

    if (wave == 0 && kbI + 1 < nkb) __builtin_amdgcn_s_wait_tensorcnt(0);
    __syncthreads();  // next buffer ready; current buffer free for overwrite
  }

#pragma unroll
  for (int r = 0; r < 8; ++r) {
    int row = qt + r + 8 * hi;
    float inv = 1.0f / lr[r];
    size_t base = ((size_t)b * N_ + row) * C_ + (size_t)h * D_;
    O[base + ln]      = f2bf(o0[r] * inv);
    O[base + 16 + ln] = f2bf(o1[r] * inv);
    O[base + 32 + ln] = f2bf(o2[r] * inv);
    O[base + 48 + ln] = f2bf(o3[r] * inv);
  }
}

extern "C" void kernel_launch(void* const* d_in, const int* in_sizes, int n_in,
                              void* d_out, int out_size, void* d_ws, size_t ws_size,
                              hipStream_t stream) {
  (void)in_sizes; (void)n_in; (void)out_size; (void)ws_size;
  const float* x     = (const float*)d_in[0];
  const float* wqkv  = (const float*)d_in[1];
  const float* wproj = (const float*)d_in[2];
  const float* bproj = (const float*)d_in[3];
  const float* lnw   = (const float*)d_in[4];
  const float* lnb   = (const float*)d_in[5];
  float* out = (float*)d_out;

  char* ws = (char*)d_ws;
  const size_t MB = 1024 * 1024;
  unsigned short* xb     = (unsigned short*)(ws + 0);        // 8 MB  x bf16 [4096,1024]
  unsigned short* wqkvT  = (unsigned short*)(ws + 8 * MB);   // 6 MB  [3072,1024]
  unsigned short* wprojT = (unsigned short*)(ws + 14 * MB);  // 2 MB  [1024,1024]
  float*          qkvf   = (float*)(ws + 16 * MB);           // 48 MB fp32 qkv
  unsigned short* attnb  = (unsigned short*)(ws + 16 * MB);  // 8 MB  (aliases consumed qkvf)
  unsigned short* qb     = (unsigned short*)(ws + 64 * MB);  // 8 MB  [B,H,N,D]
  unsigned short* kb     = (unsigned short*)(ws + 72 * MB);  // 8 MB  [B,H,N,D]
  unsigned short* vT     = (unsigned short*)(ws + 80 * MB);  // 8 MB  [B,H,D,N]

  int nx = M1 * C_;
  cvt_bf16<<<(nx + 255) / 256, 256, 0, stream>>>(x, xb, nx);
  cvt_bf16_T<<<(C_ * N3 + 255) / 256, 256, 0, stream>>>(wqkv, wqkvT, C_, N3);
  cvt_bf16_T<<<(C_ * C_ + 255) / 256, 256, 0, stream>>>(wproj, wprojT, C_, C_);

  dim3 g1(N3 / 64, M1 / 128);
  gemm_bf16<<<g1, 256, 0, stream>>>(xb, wqkvT, qkvf, nullptr, M1, N3, C_);

  qkv_prep<<<(B_ * N_ * H_ + 255) / 256, 256, 0, stream>>>(qkvf, lnw, lnb, qb, kb, vT);

  dim3 ga(B_ * H_, N_ / 64);
  attn_fwd<<<ga, 128, 0, stream>>>(qb, kb, vT, attnb);

  dim3 g2(C_ / 64, M1 / 128);
  gemm_bf16<<<g2, 256, 0, stream>>>(attnb, wprojT, out, bproj, M1, C_, C_);
}
